// KGEncoder_12111807775327
// MI455X (gfx1250) — matrix operand order repaired and checked by reading, not compiled, not run on hardware
//
#include <hip/hip_runtime.h>

// ---------------------------------------------------------------------------
// 2-layer GAT for MI455X (gfx1250, wave32).
// GEMMs run on v_wmma_f32_16x16x32_bf16 (bf16 operands, f32 accumulate).
// Weights are pre-transposed to [NC,K] bf16 so both LDS tiles stage with
// b128 loads/stores (no sub-word LDS traffic in the GEMM inner loop).
// Edge softmax/aggregation are bandwidth-bound; coalesced row-per-block
// kernels + f32/u32 global atomics.
// ---------------------------------------------------------------------------

typedef __attribute__((ext_vector_type(16))) __bf16 v16bf;
typedef __attribute__((ext_vector_type(8)))  float  v8f;

#define GAT_H     4
#define GAT_C     256      // channels per head (both layers)
#define GAT_INC   768
#define GAT_NC    1024     // H * C
#define GAT_EPS   1e-5f

// ---------------- helpers ----------------

__device__ inline __bf16 f2bf(float f) {
    // round-to-nearest-even f32 -> bf16
    unsigned u = __float_as_uint(f);
    unsigned r = (u + 0x7FFFu + ((u >> 16) & 1u)) >> 16;
    unsigned short hs = (unsigned short)r;
    __bf16 b;
    __builtin_memcpy(&b, &hs, 2);
    return b;
}

// order-preserving float <-> uint encoding for atomicMax over floats
__device__ inline unsigned fenc(float f) {
    unsigned u = __float_as_uint(f);
    return (u & 0x80000000u) ? ~u : (u | 0x80000000u);
}
__device__ inline float fdec(unsigned e) {
    unsigned u = (e & 0x80000000u) ? (e ^ 0x80000000u) : ~e;
    return __uint_as_float(u);
}

// ---------------- generic utility kernels ----------------

__global__ void fill_u32_kernel(unsigned* __restrict__ p, unsigned v, size_t n) {
    size_t i = (size_t)blockIdx.x * blockDim.x + threadIdx.x;
    if (i < n) p[i] = v;
}

__global__ void cast_f32_bf16_kernel(const float* __restrict__ in,
                                     __bf16* __restrict__ out, size_t n) {
    size_t i = (size_t)blockIdx.x * blockDim.x + threadIdx.x;
    if (i < n) out[i] = f2bf(in[i]);
}

// transpose+cast weights: in[K,NC] f32 -> out[NC,K] bf16 (tiled via LDS)
__global__ void __launch_bounds__(256)
transpose_cast_kernel(const float* __restrict__ in, __bf16* __restrict__ out,
                      int K, int NC) {
    __shared__ float tile[32][33];
    const int kBase = blockIdx.y * 32;
    const int nBase = blockIdx.x * 32;
    const int tx = threadIdx.x & 31;
    const int ty = threadIdx.x >> 5;     // 0..7
#pragma unroll
    for (int j = ty; j < 32; j += 8)
        tile[j][tx] = in[(size_t)(kBase + j) * NC + nBase + tx];
    __syncthreads();
#pragma unroll
    for (int j = ty; j < 32; j += 8)
        out[(size_t)(nBase + j) * K + kBase + tx] = f2bf(tile[tx][j]);
}

// ---------------- WMMA bf16 GEMM: C[M,NC] = A[M,K] * Bt[NC,K]^T -------------
// Workgroup = 256 threads = 8 waves; tile BM=64 x BN=128, K-step 32.
// Wave (wm, wn) computes rows wm*16..+15, cols wn*64..+63 (four 16x16 tiles,
// A fragment reused 4x -> 4 v_wmma per K-step per wave).

#define BM 64
#define BN 128
#define BK 32
#define LDSP 40   // padded LDS row stride in bf16 elements (80B, 16B-aligned)

__global__ void __launch_bounds__(256)
gemm_bf16_wmma_kernel(const __bf16* __restrict__ A, const __bf16* __restrict__ Bt,
                      float* __restrict__ C, int M, int K, int NC) {
    __shared__ __bf16 As[BM][LDSP];      // As[m][k]
    __shared__ __bf16 Bs[BN][LDSP];      // Bs[n][k]  (Bt is already [NC,K])

    const int tid   = threadIdx.x;
    const int lane  = tid & 31;
    const int wave  = tid >> 5;
    const int waveM = wave >> 1;       // 0..3
    const int waveN = wave & 1;        // 0..1
    const int hi    = lane >> 4;       // lane group (0/1)
    const int lm    = lane & 15;

    const int rowBlock = blockIdx.y * BM;
    const int colBlock = blockIdx.x * BN;

    v8f acc[4] = {v8f{}, v8f{}, v8f{}, v8f{}};

    // global -> LDS staging indices
    const int aRow = tid >> 2;           // 0..63
    const int aSeg = (tid & 3) * 8;      // 0,8,16,24  (bf16 elems)
    const int bRow0 = tid >> 2;          // 0..63      (uint4 u = tid)
    const int bRow1 = (tid + 256) >> 2;  // 64..127    (uint4 u = tid+256)
    const int bSeg  = (tid & 3) * 8;

    for (int k0 = 0; k0 < K; k0 += BK) {
        // stage A (row-major activations) and B (pre-transposed weights)
        uint4 av = make_uint4(0u, 0u, 0u, 0u);
        const int gr = rowBlock + aRow;
        if (gr < M) av = *(const uint4*)(A + (size_t)gr * K + k0 + aSeg);
        const uint4 bv0 = *(const uint4*)(Bt + (size_t)(colBlock + bRow0) * K + k0 + bSeg);
        const uint4 bv1 = *(const uint4*)(Bt + (size_t)(colBlock + bRow1) * K + k0 + bSeg);
        if (k0 + BK < K) {
            __builtin_prefetch(Bt + (size_t)(colBlock + bRow0) * K + k0 + BK + bSeg, 0, 1);
            __builtin_prefetch(Bt + (size_t)(colBlock + bRow1) * K + k0 + BK + bSeg, 0, 1);
        }
        __syncthreads();   // previous iteration's fragment reads done
        *(uint4*)&As[aRow][aSeg]  = av;
        *(uint4*)&Bs[bRow0][bSeg] = bv0;
        *(uint4*)&Bs[bRow1][bSeg] = bv1;
        __syncthreads();

        // A fragment (16x32 bf16, ISA layout):
        //  lanes 0-15 row m: halves0-7 = K 0..7,  halves8-15 = K 16..23
        //  lanes16-31 row m: halves0-7 = K 8..15, halves8-15 = K 24..31
        v16bf af;
        const __bf16* ap = &As[waveM * 16 + lm][hi * 8];
        ((uint4*)&af)[0] = *(const uint4*)ap;
        ((uint4*)&af)[1] = *(const uint4*)(ap + 16);

        // B fragments: col = lane%16, same K order as A (halves0-7 = K hi*8.., halves8-15 = K 16+hi*8..)
#pragma unroll
        for (int n = 0; n < 4; ++n) {
            v16bf bfr;
            const __bf16* bp = &Bs[waveN * 64 + n * 16 + lm][hi * 8];
            ((uint4*)&bfr)[0] = *(const uint4*)bp;
            ((uint4*)&bfr)[1] = *(const uint4*)(bp + 16);
            acc[n] = __builtin_amdgcn_wmma_f32_16x16x32_bf16(
                false, af, false, bfr, (short)0, acc[n], false, false);
        }
    }

    // D layout: lane group hi -> rows hi*8 + j (VGPR j), col = lane%16
    const int rbase = rowBlock + waveM * 16 + hi * 8;
#pragma unroll
    for (int n = 0; n < 4; ++n) {
        const int col = colBlock + waveN * 64 + n * 16 + lm;
#pragma unroll
        for (int j = 0; j < 8; ++j) {
            const int r = rbase + j;
            if (r < M) C[(size_t)r * NC + col] = acc[n][j];
        }
    }
}

// ---------------- attention score dots: s/d[n,h] = <xh[n,h,:], a[h,:]> -------
// one wave per (node,head); xh flat offset = (n*H+h)*C = idx*C

__global__ void __launch_bounds__(256)
attn_scores_kernel(const float* __restrict__ xh, const float* __restrict__ asrc,
                   const float* __restrict__ adst, float* __restrict__ sOut,
                   float* __restrict__ dOut, int NH) {
    const int idx  = blockIdx.x * 8 + (threadIdx.x >> 5);
    const int lane = threadIdx.x & 31;
    if (idx >= NH) return;
    const int head = idx & (GAT_H - 1);
    const float* row = xh + (size_t)idx * GAT_C;
    const float* va  = asrc + head * GAT_C;
    const float* vd  = adst + head * GAT_C;
    float ss = 0.f, dd = 0.f;
    for (int j = lane; j < GAT_C; j += 32) {
        const float v = row[j];
        ss += v * va[j];
        dd += v * vd[j];
    }
#pragma unroll
    for (int o = 16; o > 0; o >>= 1) {
        ss += __shfl_xor(ss, o, 32);
        dd += __shfl_xor(dd, o, 32);
    }
    if (lane == 0) { sOut[idx] = ss; dOut[idx] = dd; }
}

// ---------------- edge kernels (self-loops appended at i >= E) --------------

__global__ void edge_max_kernel(const int* __restrict__ esrc, const int* __restrict__ edst,
                                const float* __restrict__ sv, const float* __restrict__ dv,
                                float* __restrict__ ebuf, unsigned* __restrict__ emax,
                                int E, int N) {
    const int i = blockIdx.x * blockDim.x + threadIdx.x;
    if (i >= E + N) return;
    int si, di;
    if (i < E) { si = esrc[i]; di = edst[i]; } else { si = di = i - E; }
#pragma unroll
    for (int h = 0; h < GAT_H; ++h) {
        float e = sv[si * GAT_H + h] + dv[di * GAT_H + h];
        e = (e > 0.f) ? e : 0.2f * e;               // leaky_relu(0.2)
        ebuf[(size_t)i * GAT_H + h] = e;
        atomicMax(&emax[di * GAT_H + h], fenc(e));
    }
}

__global__ void edge_exp_kernel(const int* __restrict__ edst,
                                float* __restrict__ ebuf,
                                const unsigned* __restrict__ emax,
                                float* __restrict__ denom, int E, int N) {
    const int i = blockIdx.x * blockDim.x + threadIdx.x;
    if (i >= E + N) return;
    const int di = (i < E) ? edst[i] : (i - E);
#pragma unroll
    for (int h = 0; h < GAT_H; ++h) {
        const float m  = fdec(emax[di * GAT_H + h]);
        const float ex = __expf(ebuf[(size_t)i * GAT_H + h] - m);
        ebuf[(size_t)i * GAT_H + h] = ex;
        atomicAdd(&denom[di * GAT_H + h], ex);
    }
}

// one 256-thread block per edge: agg[dst] += alpha * xh[src]  (coalesced rows)
__global__ void __launch_bounds__(256)
aggregate_kernel(const int* __restrict__ esrc, const int* __restrict__ edst,
                 const float* __restrict__ ebuf, const float* __restrict__ denom,
                 const float* __restrict__ xh, float* __restrict__ agg,
                 int E, int N) {
    const int i = blockIdx.x;
    int si, di;
    if (i < E) { si = esrc[i]; di = edst[i]; } else { si = di = i - E; }
    __shared__ float alpha[GAT_H];
    if (threadIdx.x < GAT_H) {
        alpha[threadIdx.x] = ebuf[(size_t)i * GAT_H + threadIdx.x] /
                             denom[di * GAT_H + threadIdx.x];
    }
    __syncthreads();
    const int t = threadIdx.x;
    const float* srow = xh + (size_t)si * GAT_NC;
    float* drow       = agg + (size_t)di * GAT_NC;
#pragma unroll
    for (int h = 0; h < GAT_H; ++h) {
        atomicAdd(&drow[h * GAT_C + t], alpha[h] * srow[h * GAT_C + t]);
    }
}

// out[n,c] = mean_h agg[n,h,c] + bias[c]
__global__ void head_mean_bias_kernel(const float* __restrict__ agg,
                                      const float* __restrict__ bias,
                                      float* __restrict__ out, int N) {
    const int id = blockIdx.x * 256 + threadIdx.x;
    if (id >= N * GAT_C) return;
    const int n = id >> 8, c = id & 255;
    const float* r = agg + (size_t)n * GAT_NC;
    out[id] = 0.25f * (r[c] + r[GAT_C + c] + r[2 * GAT_C + c] + r[3 * GAT_C + c]) + bias[c];
}

// ---------------- BatchNorm (batch statistics) + ReLU + bf16 cast -----------

__global__ void __launch_bounds__(256)
bn_stats_kernel(const float* __restrict__ h, float* __restrict__ sums, int N) {
    const int c = threadIdx.x;            // 256 channels
    float s = 0.f, s2 = 0.f;
    for (int r = blockIdx.x; r < N; r += gridDim.x) {
        const float v = h[(size_t)r * GAT_C + c];
        s += v; s2 += v * v;
    }
    atomicAdd(&sums[c], s);
    atomicAdd(&sums[GAT_C + c], s2);
}

__global__ void bn_apply_relu_cast_kernel(const float* __restrict__ h,
                                          const float* __restrict__ sums,
                                          const float* __restrict__ gamma,
                                          const float* __restrict__ beta,
                                          __bf16* __restrict__ outb, int N) {
    const int id = blockIdx.x * 256 + threadIdx.x;
    if (id >= N * GAT_C) return;
    const int c = id & 255;
    const float invn = 1.0f / (float)N;
    const float mu  = sums[c] * invn;
    const float var = sums[GAT_C + c] * invn - mu * mu;
    float v = (h[id] - mu) * __frsqrt_rn(var + GAT_EPS) * gamma[c] + beta[c];
    v = fmaxf(v, 0.f);
    outb[id] = f2bf(v);
}

// ---------------------------------------------------------------------------
// host-side launcher
// ---------------------------------------------------------------------------

extern "C" void kernel_launch(void* const* d_in, const int* in_sizes, int n_in,
                              void* d_out, int out_size, void* d_ws, size_t ws_size,
                              hipStream_t stream) {
    const float* x     = (const float*)d_in[0];
    const int*   edges = (const int*)d_in[1];     // [2,E]: src row then dst row
    const float* W1    = (const float*)d_in[2];
    const float* as1   = (const float*)d_in[3];
    const float* ad1   = (const float*)d_in[4];
    const float* b1    = (const float*)d_in[5];
    const float* gamma = (const float*)d_in[6];
    const float* beta  = (const float*)d_in[7];
    const float* W2    = (const float*)d_in[8];
    const float* as2   = (const float*)d_in[9];
    const float* ad2   = (const float*)d_in[10];
    const float* b2    = (const float*)d_in[11];

    const int N = in_sizes[0] / GAT_INC;
    const int E = in_sizes[1] / 2;
    const int* esrc = edges;
    const int* edst = edges + E;
    const int TOT = E + N;                 // edges + self loops

    // ---- carve workspace ----
    char* ws = (char*)d_ws;
    size_t off = 0;
    auto carve = [&](size_t bytes) -> void* {
        void* p = ws + off;
        off += (bytes + 255) & ~(size_t)255;
        return p;
    };
    __bf16*   xb     = (__bf16*)carve((size_t)N * GAT_INC * 2);
    __bf16*   w1t    = (__bf16*)carve((size_t)GAT_INC * GAT_NC * 2);  // [NC, INC]
    __bf16*   w2t    = (__bf16*)carve((size_t)GAT_C * GAT_NC * 2);    // [NC, C]
    __bf16*   hb     = (__bf16*)carve((size_t)N * GAT_C * 2);
    float*    xh     = (float*)carve((size_t)N * GAT_NC * 4);
    float*    agg    = (float*)carve((size_t)N * GAT_NC * 4);
    float*    hfeat  = (float*)carve((size_t)N * GAT_C * 4);
    float*    sval   = (float*)carve((size_t)N * GAT_H * 4);
    float*    dval   = (float*)carve((size_t)N * GAT_H * 4);
    unsigned* emax   = (unsigned*)carve((size_t)N * GAT_H * 4);
    float*    denom  = (float*)carve((size_t)N * GAT_H * 4);
    float*    ebuf   = (float*)carve((size_t)TOT * GAT_H * 4);
    float*    bnsums = (float*)carve(2 * GAT_C * 4);

    auto g1 = [](size_t n) { return (unsigned)((n + 255) / 256); };

    // ---- casts: activations row-major, weights transposed to [NC,K] ----
    cast_f32_bf16_kernel<<<g1((size_t)N * GAT_INC), 256, 0, stream>>>(x, xb, (size_t)N * GAT_INC);
    transpose_cast_kernel<<<dim3(GAT_NC / 32, GAT_INC / 32), 256, 0, stream>>>(W1, w1t, GAT_INC, GAT_NC);
    transpose_cast_kernel<<<dim3(GAT_NC / 32, GAT_C / 32), 256, 0, stream>>>(W2, w2t, GAT_C, GAT_NC);

    const dim3 gemmGrid(GAT_NC / BN, (N + BM - 1) / BM);

    for (int layer = 0; layer < 2; ++layer) {
        // zero accumulators
        fill_u32_kernel<<<g1((size_t)N * GAT_NC), 256, 0, stream>>>((unsigned*)agg, 0u, (size_t)N * GAT_NC);
        fill_u32_kernel<<<g1((size_t)N * GAT_H), 256, 0, stream>>>(emax, 0u, (size_t)N * GAT_H);
        fill_u32_kernel<<<g1((size_t)N * GAT_H), 256, 0, stream>>>((unsigned*)denom, 0u, (size_t)N * GAT_H);

        // xh = A @ W  (WMMA bf16)
        if (layer == 0)
            gemm_bf16_wmma_kernel<<<gemmGrid, 256, 0, stream>>>(xb, w1t, xh, N, GAT_INC, GAT_NC);
        else
            gemm_bf16_wmma_kernel<<<gemmGrid, 256, 0, stream>>>(hb, w2t, xh, N, GAT_C, GAT_NC);

        // attention scores
        const float* pas = (layer == 0) ? as1 : as2;
        const float* pad = (layer == 0) ? ad1 : ad2;
        attn_scores_kernel<<<g1((size_t)N * GAT_H * 32), 256, 0, stream>>>(xh, pas, pad, sval, dval, N * GAT_H);

        // segment softmax + weighted aggregation
        edge_max_kernel<<<g1((size_t)TOT), 256, 0, stream>>>(esrc, edst, sval, dval, ebuf, emax, E, N);
        edge_exp_kernel<<<g1((size_t)TOT), 256, 0, stream>>>(edst, ebuf, emax, denom, E, N);
        aggregate_kernel<<<TOT, 256, 0, stream>>>(esrc, edst, ebuf, denom, xh, agg, E, N);

        if (layer == 0) {
            head_mean_bias_kernel<<<g1((size_t)N * GAT_C), 256, 0, stream>>>(agg, b1, hfeat, N);
            // BatchNorm (batch stats) + ReLU, cast to bf16 for layer-2 GEMM
            fill_u32_kernel<<<g1(2 * GAT_C), 256, 0, stream>>>((unsigned*)bnsums, 0u, 2 * GAT_C);
            bn_stats_kernel<<<64, 256, 0, stream>>>(hfeat, bnsums, N);
            bn_apply_relu_cast_kernel<<<g1((size_t)N * GAT_C), 256, 0, stream>>>(hfeat, bnsums, gamma, beta, hb, N);
        } else {
            head_mean_bias_kernel<<<g1((size_t)N * GAT_C), 256, 0, stream>>>(agg, b2, (float*)d_out, N);
        }
    }
    (void)n_in; (void)out_size; (void)ws_size;
}
